// ConvAttention_83356725281304
// MI455X (gfx1250) — compile-verified, matrix operand order checked by reading, not run on hardware
//
#include <hip/hip_runtime.h>
#include <hip/hip_bf16.h>
#include <math.h>

// ---------------------------------------------------------------------------
// ConvAttention for MI455X (gfx1250, wave32, WMMA).
// All GEMMs use v_wmma_f32_16x16x32_f16 (f16 operands, f32 accum).
//
// Data layouts:
//   activations: [B][Tpad][C] f16, with `pad` zero rows top+bottom for k=3.
//   weights:     [Cout][KW*Cin] f16 with K = dk*Cin + ci (tap-outer) so that
//                an im2col row for output position t is CONTIGUOUS memory
//                starting at padded row t (row stride == Cin).
// WMMA fragment layouts per CDNA5 ISA 7.12.2:
//   A (16x32 f16): lane holds M=lane&15; halves = K runs {koff+0..7,koff+16..23}
//                  (lanes 0-15) / {koff+8..15,koff+24..31} (lanes 16-31).
//   B (32x16 f16): mirrored, lane holds N=lane&15.
//   C/D (16x16 f32): VGPR r, lane l -> M = r + 8*(l>=16), N = l&15.
// ---------------------------------------------------------------------------

typedef __attribute__((ext_vector_type(16))) _Float16 v16h;
typedef __attribute__((ext_vector_type(8)))  _Float16 v8h;
typedef __attribute__((ext_vector_type(8)))  float    v8f;
typedef __attribute__((ext_vector_type(4)))  float    v4f;

__device__ __forceinline__ v8h zero_v8h() {
  v8h z = { (_Float16)0, (_Float16)0, (_Float16)0, (_Float16)0,
            (_Float16)0, (_Float16)0, (_Float16)0, (_Float16)0 };
  return z;
}

__device__ __forceinline__ v8f wmma_f16(v16h a, v16h b, v8f c) {
  // (neg_a, A, neg_b, B, c_mod, C, reuse_a, reuse_b)
  return __builtin_amdgcn_wmma_f32_16x16x32_f16(false, a, false, b,
                                                (short)0, c, false, false);
}

// Load one 16x32 (or mirrored 32x16) f16 fragment whose K-row is contiguous
// at `row`. Runs of 8 halves are 16B-aligned b128 loads; runs with K >= KTOT
// are zeroed (handles K=80 -> padded-to-96 and K=240 -> padded-to-256).
template<int KTOT>
__device__ __forceinline__ v16h load_frag(const _Float16* __restrict__ row,
                                          int koff, int lane) {
  const int base = koff + ((lane & 16) ? 8 : 0);
  v8h lo = (base      < KTOT) ? *(const v8h*)(row + base)      : zero_v8h();
  v8h hi = (base + 16 < KTOT) ? *(const v8h*)(row + base + 16) : zero_v8h();
  v16h r;
#pragma unroll
  for (int i = 0; i < 8; ++i) { r[i] = lo[i]; r[i + 8] = hi[i]; }
  return r;
}

// ---------------------------------------------------------------------------
// Packing kernels (small, bandwidth-trivial; everything lives in L2 after).
// ---------------------------------------------------------------------------

// [B][C][T] f32 -> [B][T+2][C] f16 with zero rows at p=0 and p=T+1.
__global__ void pack_transpose_kernel(const float* __restrict__ in,
                                      _Float16* __restrict__ out,
                                      int B, int C, int T) {
  size_t total = (size_t)B * (T + 2) * C;
  for (size_t i = (size_t)blockIdx.x * blockDim.x + threadIdx.x; i < total;
       i += (size_t)gridDim.x * blockDim.x) {
    int c = (int)(i % C);
    size_t r = i / C;
    int p = (int)(r % (T + 2));
    int b = (int)(r / (T + 2));
    float v = 0.f;
    if (p >= 1 && p <= T) v = in[((size_t)b * C + c) * T + (p - 1)];
    out[i] = (_Float16)v;
  }
}

// [CO][CIN][3] f32 -> [CO][3][CIN] f16 (tap-outer K ordering).
__global__ void pack_w3_kernel(const float* __restrict__ in,
                               _Float16* __restrict__ out, int CO, int CIN) {
  int total = CO * 3 * CIN;
  for (int i = blockIdx.x * blockDim.x + threadIdx.x; i < total;
       i += gridDim.x * blockDim.x) {
    int co  = i / (3 * CIN);
    int rem = i - co * 3 * CIN;
    int dk  = rem / CIN;
    int ci  = rem - dk * CIN;
    out[i] = (_Float16)in[((size_t)co * CIN + ci) * 3 + dk];
  }
}

__global__ void cvt_f32_f16_kernel(const float* __restrict__ in,
                                   _Float16* __restrict__ out, int n) {
  int i = blockIdx.x * blockDim.x + threadIdx.x;
  if (i < n) out[i] = (_Float16)in[i];
}

// Per-row sum of squares over 80 channels (rows of [R][80] f16).
__global__ void sumsq80_kernel(const _Float16* __restrict__ x,
                               float* __restrict__ o, int rows) {
  int r = blockIdx.x * blockDim.x + threadIdx.x;
  if (r >= rows) return;
  const _Float16* p = x + (size_t)r * 80;
  float s = 0.f;
#pragma unroll
  for (int c = 0; c < 80; ++c) { float v = (float)p[c]; s += v * v; }
  o[r] = s;
}

// ---------------------------------------------------------------------------
// Fused conv-as-GEMM: out[b][t][co] = relu?(bias[co] + sum_K W[co,K]*P[t,K])
// Block tile: WPB waves x 16 Cout rows each, 64 output positions.
// Each wave: 4 WMMA accumulators (16Mx64N), K loop in steps of 32.
// ---------------------------------------------------------------------------
template<int CO, int KTOT, int CIN, bool RELU, int WPB>
__global__ void conv_gemm_kernel(const _Float16* __restrict__ inp,  // [B][Tpad][CIN]
                                 const _Float16* __restrict__ wgt,  // [CO][KTOT]
                                 const float*    __restrict__ bias, // [CO]
                                 _Float16*       __restrict__ out,  // [B][T][CO]
                                 int T, int Tpad, int mblocks) {
  const int lane = threadIdx.x & 31;
  const int wave = threadIdx.x >> 5;
  const int mb = blockIdx.x % mblocks;
  const int nb = blockIdx.x / mblocks;
  const int n0 = nb * 64;
  const int b  = n0 / T;
  const int t0 = n0 - b * T;
  const int m16 = lane & 15;
  const int co0 = (mb * WPB + wave) * 16;

  const _Float16* wrow = wgt + (size_t)(co0 + m16) * KTOT;
  const _Float16* pb   = inp + ((size_t)b * Tpad + (t0 + m16)) * CIN;
  const size_t nstep = (size_t)16 * CIN;   // 16 positions per N-tile

  v8f c0 = {}, c1 = {}, c2 = {}, c3 = {};
  for (int koff = 0; koff < KTOT; koff += 32) {
    v16h a  = load_frag<KTOT>(wrow, koff, lane);
    v16h b0 = load_frag<KTOT>(pb,             koff, lane);
    v16h b1 = load_frag<KTOT>(pb +     nstep, koff, lane);
    v16h b2 = load_frag<KTOT>(pb + 2 * nstep, koff, lane);
    v16h b3 = load_frag<KTOT>(pb + 3 * nstep, koff, lane);
    c0 = wmma_f16(a, b0, c0);
    c1 = wmma_f16(a, b1, c1);
    c2 = wmma_f16(a, b2, c2);
    c3 = wmma_f16(a, b3, c3);
  }

  const int hi  = lane >> 4;
  const int cob = co0 + 8 * hi;
  v4f blo = *(const v4f*)(bias + cob);
  v4f bhi = *(const v4f*)(bias + cob + 4);
  float bs[8] = { blo[0], blo[1], blo[2], blo[3], bhi[0], bhi[1], bhi[2], bhi[3] };

  auto emit = [&](v8f c, int j) {
    const int t = t0 + j * 16 + m16;
    v8h h;
#pragma unroll
    for (int r = 0; r < 8; ++r) {
      float v = c[r] + bs[r];
      if (RELU) v = v > 0.f ? v : 0.f;
      h[r] = (_Float16)v;
    }
    *(v8h*)(out + ((size_t)b * T + t) * CO + cob) = h;
  };
  emit(c0, 0); emit(c1, 1); emit(c2, 2); emit(c3, 3);
}

// ---------------------------------------------------------------------------
// Fused distance + log_softmax + prior + masked softmax.
// One block = 1 batch x 32 tm rows x all 256 tt columns.
// 8 waves: wave w -> M-tile (w>>2), 64-column strip ((w&3)*64) -> 4 WMMA accs.
// G = q.k goes to LDS; logit = -5e-4*(q2 + k2 - 2G).
// ---------------------------------------------------------------------------
__global__ __launch_bounds__(256) void attn_softmax_kernel(
    const _Float16* __restrict__ q3T,   // [B][1024][80]
    const _Float16* __restrict__ kT,    // [B][256][80]
    const float* __restrict__ q2s,      // [B*1024]
    const float* __restrict__ k2s,      // [B*256]
    const float* __restrict__ prior,    // [B][1024][256]
    const unsigned char* __restrict__ mask, // [B][256] (bool)
    float* __restrict__ out_attn,       // [B][1024][256]
    float* __restrict__ out_logp) {     // [B][1024][256]
  constexpr int TT = 256, ROWS = 32, LSTR = 260;  // 260: bank-conflict pad
  __shared__ float Ls[ROWS][LSTR];
  __shared__ float k2sh[TT];
  __shared__ float msk[TT];

  const int b   = blockIdx.x >> 5;   // 32 row-blocks per batch
  const int rb  = blockIdx.x & 31;
  const int tm0 = rb * ROWS;
  const int tid = threadIdx.x;
  const int lane = tid & 31, wave = tid >> 5;

  if (tid < TT) {
    k2sh[tid] = k2s[(size_t)b * TT + tid];
    msk[tid]  = mask[(size_t)b * TT + tid] ? 1.f : 0.f;
  }

  // ---- GEMM phase: G[tm][tt] = sum_c q3[tm,c]*k[tt,c], K=80 padded to 96
  const int m16 = lane & 15, hi = lane >> 4;
  const int mt = wave >> 2;            // 0..1 -> rows mt*16..mt*16+15
  const int nq = (wave & 3) * 64;      // column strip
  const _Float16* arow = q3T + ((size_t)b * 1024 + tm0 + mt * 16 + m16) * 80;
  const _Float16* brow = kT  + ((size_t)b * TT + nq + m16) * 80;

  v8f acc[4];
#pragma unroll
  for (int j = 0; j < 4; ++j) { v8f z = {}; acc[j] = z; }
  for (int koff = 0; koff < 80; koff += 32) {
    v16h a = load_frag<80>(arow, koff, lane);
#pragma unroll
    for (int j = 0; j < 4; ++j) {
      v16h bf = load_frag<80>(brow + (size_t)j * 16 * 80, koff, lane);
      acc[j] = wmma_f16(a, bf, acc[j]);
    }
  }
#pragma unroll
  for (int j = 0; j < 4; ++j)
#pragma unroll
    for (int r = 0; r < 8; ++r)
      Ls[mt * 16 + 8 * hi + r][nq + j * 16 + m16] = acc[j][r];
  __syncthreads();

  // ---- Softmax phase: 8 threads per row, columns c = e + 8*i (no bank conflict)
  const int row = tid >> 3;
  const int e   = tid & 7;
  const int tm  = tm0 + row;
  const float q2 = q2s[(size_t)b * 1024 + tm];
  const float* prow = prior   + ((size_t)b * 1024 + tm) * TT;
  float* lpout      = out_logp + ((size_t)b * 1024 + tm) * TT;
  float* atout      = out_attn + ((size_t)b * 1024 + tm) * TT;

  float mx = -INFINITY;
#pragma unroll 4
  for (int i = 0; i < 32; ++i) {
    int c = e + 8 * i;
    float logit = -5.0e-4f * (q2 + k2sh[c] - 2.f * Ls[row][c]);
    mx = fmaxf(mx, logit);
  }
  mx = fmaxf(mx, __shfl_xor(mx, 1));
  mx = fmaxf(mx, __shfl_xor(mx, 2));
  mx = fmaxf(mx, __shfl_xor(mx, 4));

  float s = 0.f;
#pragma unroll 4
  for (int i = 0; i < 32; ++i) {
    int c = e + 8 * i;
    float logit = -5.0e-4f * (q2 + k2sh[c] - 2.f * Ls[row][c]);
    s += __expf(logit - mx);
  }
  s += __shfl_xor(s, 1); s += __shfl_xor(s, 2); s += __shfl_xor(s, 4);
  const float L = mx + __logf(s);   // logsumexp

  float mx2 = -INFINITY;
#pragma unroll 4
  for (int i = 0; i < 32; ++i) {
    int c = e + 8 * i;
    float logit = -5.0e-4f * (q2 + k2sh[c] - 2.f * Ls[row][c]);
    float lp = logit - L + __logf(prow[c] + 1e-8f);
    lpout[c] = lp;
    float v = (msk[c] != 0.f) ? lp : -INFINITY;
    Ls[row][c] = v;                 // each thread rewrites only its own cells
    mx2 = fmaxf(mx2, v);
  }
  mx2 = fmaxf(mx2, __shfl_xor(mx2, 1));
  mx2 = fmaxf(mx2, __shfl_xor(mx2, 2));
  mx2 = fmaxf(mx2, __shfl_xor(mx2, 4));
  if (mx2 == -INFINITY) mx2 = 0.f;

  float s2 = 0.f;
#pragma unroll 4
  for (int i = 0; i < 32; ++i) s2 += __expf(Ls[row][e + 8 * i] - mx2);
  s2 += __shfl_xor(s2, 1); s2 += __shfl_xor(s2, 2); s2 += __shfl_xor(s2, 4);
  const float inv = 1.f / s2;
#pragma unroll 4
  for (int i = 0; i < 32; ++i) {
    int c = e + 8 * i;
    atout[c] = __expf(Ls[row][c] - mx2) * inv;
  }
}

// ---------------------------------------------------------------------------
// Host launcher
// ---------------------------------------------------------------------------
extern "C" void kernel_launch(void* const* d_in, const int* in_sizes, int n_in,
                              void* d_out, int out_size, void* d_ws, size_t ws_size,
                              hipStream_t stream) {
  (void)in_sizes; (void)n_in; (void)out_size; (void)ws_size;
  const int B = 8, Cmel = 80, Ctxt = 512, Tm = 1024, Tt = 256;

  const float* text  = (const float*)d_in[0];
  const float* mels  = (const float*)d_in[1];
  const unsigned char* mask = (const unsigned char*)d_in[2]; // jnp bool = 1 byte
  const float* prior = (const float*)d_in[3];
  const float* kw1 = (const float*)d_in[4];
  const float* kb1 = (const float*)d_in[5];
  const float* kw2 = (const float*)d_in[6];
  const float* kb2 = (const float*)d_in[7];
  const float* qw1 = (const float*)d_in[8];
  const float* qb1 = (const float*)d_in[9];
  const float* qw2 = (const float*)d_in[10];
  const float* qb2 = (const float*)d_in[11];
  const float* qw3 = (const float*)d_in[12];
  const float* qb3 = (const float*)d_in[13];

  char* ws = (char*)d_ws;
  size_t off = 0;
  auto alloc = [&](size_t bytes) -> char* {
    char* p = ws + off;
    off = (off + bytes + 255) & ~(size_t)255;
    return p;
  };
  _Float16* textT = (_Float16*)alloc((size_t)B * (Tt + 2) * Ctxt * 2);
  _Float16* melsT = (_Float16*)alloc((size_t)B * (Tm + 2) * Cmel * 2);
  _Float16* w1p   = (_Float16*)alloc((size_t)1024 * 1536 * 2);
  _Float16* w2p   = (_Float16*)alloc((size_t)80 * 1024 * 2);
  _Float16* w3p   = (_Float16*)alloc((size_t)160 * 240 * 2);
  _Float16* w4p   = (_Float16*)alloc((size_t)80 * 160 * 2);
  _Float16* w5p   = (_Float16*)alloc((size_t)80 * 80 * 2);
  _Float16* h1T   = (_Float16*)alloc((size_t)B * Tt * 1024 * 2);
  _Float16* q1T   = (_Float16*)alloc((size_t)B * Tm * 160 * 2);
  _Float16* q2T   = (_Float16*)alloc((size_t)B * Tm * 80 * 2);
  _Float16* q3T   = (_Float16*)alloc((size_t)B * Tm * 80 * 2);
  _Float16* kTn   = (_Float16*)alloc((size_t)B * Tt * 80 * 2);
  float*    k2s   = (float*)alloc((size_t)B * Tt * 4);
  float*    q2s   = (float*)alloc((size_t)B * Tm * 4);

  float* out_attn = (float*)d_out;
  float* out_logp = out_attn + (size_t)B * Tm * Tt;

  auto cdiv = [](long a, long b) { return (int)((a + b - 1) / b); };

  // --- pack / convert ---
  pack_transpose_kernel<<<cdiv((long)B * (Tt + 2) * Ctxt, 256), 256, 0, stream>>>(
      text, textT, B, Ctxt, Tt);
  pack_transpose_kernel<<<cdiv((long)B * (Tm + 2) * Cmel, 256), 256, 0, stream>>>(
      mels, melsT, B, Cmel, Tm);
  pack_w3_kernel<<<cdiv(1024L * 3 * 512, 256), 256, 0, stream>>>(kw1, w1p, 1024, 512);
  pack_w3_kernel<<<cdiv(160L * 3 * 80, 256), 256, 0, stream>>>(qw1, w3p, 160, 80);
  cvt_f32_f16_kernel<<<cdiv(80L * 1024, 256), 256, 0, stream>>>(kw2, w2p, 80 * 1024);
  cvt_f32_f16_kernel<<<cdiv(80L * 160, 256), 256, 0, stream>>>(qw2, w4p, 80 * 160);
  cvt_f32_f16_kernel<<<cdiv(80L * 80, 256), 256, 0, stream>>>(qw3, w5p, 80 * 80);

  // --- key encoder: conv(512->1024,k3)+ReLU ; conv(1024->80,k1) ---
  conv_gemm_kernel<1024, 1536, 512, true, 4>
      <<<16 * (B * Tt / 64), 128, 0, stream>>>(textT, w1p, kb1, h1T, Tt, Tt + 2, 16);
  conv_gemm_kernel<80, 1024, 1024, false, 5>
      <<<1 * (B * Tt / 64), 160, 0, stream>>>(h1T, w2p, kb2, kTn, Tt, Tt, 1);

  // --- query encoder: conv(80->160,k3)+ReLU ; conv(160->80)+ReLU ; conv(80->80) ---
  conv_gemm_kernel<160, 240, 80, true, 5>
      <<<2 * (B * Tm / 64), 160, 0, stream>>>(melsT, w3p, qb1, q1T, Tm, Tm + 2, 2);
  conv_gemm_kernel<80, 160, 160, true, 5>
      <<<1 * (B * Tm / 64), 160, 0, stream>>>(q1T, w4p, qb2, q2T, Tm, Tm, 1);
  conv_gemm_kernel<80, 80, 80, false, 5>
      <<<1 * (B * Tm / 64), 160, 0, stream>>>(q2T, w5p, qb3, q3T, Tm, Tm, 1);

  // --- squared norms ---
  sumsq80_kernel<<<cdiv((long)B * Tt, 256), 256, 0, stream>>>(kTn, k2s, B * Tt);
  sumsq80_kernel<<<cdiv((long)B * Tm, 256), 256, 0, stream>>>(q3T, q2s, B * Tm);

  // --- fused distance + softmax / logprob ---
  attn_softmax_kernel<<<B * (Tm / 32), 256, 0, stream>>>(
      q3T, kTn, q2s, k2s, prior, mask, out_attn, out_logp);
}
